// Block_60662118089108
// MI455X (gfx1250) — matrix operand order repaired and checked by reading, not compiled, not run on hardware
//
#include <hip/hip_runtime.h>

#define DEV __device__ __forceinline__

typedef __attribute__((ext_vector_type(16))) __bf16 v16bf;
typedef __attribute__((ext_vector_type(8)))  __bf16 v8bf;
typedef __attribute__((ext_vector_type(8)))  float  v8f;
typedef __attribute__((ext_vector_type(4)))  float  v4f;

constexpr int Bc = 2, Tc = 1024, Cc = 2048;
constexpr int Hc = 32, KVc = 8, Dc = 128;
constexpr int Ec = 8, Fc = 768;
constexpr int NTOK = Bc * Tc;   // 2048 tokens

// LDS tile geometry: rows padded to 80B (40 bf16) so strided fragment reads
// land on distinct banks (20*ln mod 64 is distinct for ln=0..15).
constexpr int LROW = 40;        // bf16 units per padded row (80 bytes)

// ---------------- bf16 helpers (bit ops only) --------------------------------
DEV unsigned short bfbits(float f) {
  union { float f; unsigned u; } x; x.f = f;
  unsigned r = x.u + 0x7fffu + ((x.u >> 16) & 1u);   // RNE
  return (unsigned short)(r >> 16);
}
DEV __bf16 f2bf(float f) { return __builtin_bit_cast(__bf16, bfbits(f)); }
DEV float bf2f(__bf16 b) {
  union { unsigned u; float f; } x;
  x.u = ((unsigned)__builtin_bit_cast(unsigned short, b)) << 16;
  return x.f;
}

// ---------------- CDNA5 async global->LDS (gfx1250) --------------------------
DEV void async_copy_b128(unsigned lds_byte_addr, const void* gaddr) {
  asm volatile("global_load_async_to_lds_b128 %0, %1, off"
               :: "v"(lds_byte_addr), "v"(gaddr) : "memory");
}
DEV void wait_asynccnt0() { asm volatile("s_wait_asynccnt 0x0" ::: "memory"); }

// ---------------- WMMA fragment loaders (wave32, contiguous k order in A and B) -------
// A fragment 16x32 bf16 from a padded LDS tile: row = lane&15,
// pair i covers k = 2i + 16*(lane>=16)
DEV v16bf load_a_frag_lds(const __bf16* aT, int ln, int hi) {
  v16bf a;
#pragma unroll
  for (int i = 0; i < 8; ++i) {
    int kb = 2 * i + hi * 16;
    unsigned p = *(const unsigned*)(aT + ln * LROW + kb);
    a[2 * i]     = __builtin_bit_cast(__bf16, (unsigned short)(p & 0xffffu));
    a[2 * i + 1] = __builtin_bit_cast(__bf16, (unsigned short)(p >> 16));
  }
  return a;
}
// A fragment straight from global row-major bf16 (used by attention)
DEV v16bf load_a_frag_g(const __bf16* rowptr, int k0, int hi) {
  v16bf a;
#pragma unroll
  for (int i = 0; i < 8; ++i) {
    int kb = 2 * i + hi * 16;
    unsigned p = *(const unsigned*)(rowptr + k0 + kb);
    a[2 * i]     = __builtin_bit_cast(__bf16, (unsigned short)(p & 0xffffu));
    a[2 * i + 1] = __builtin_bit_cast(__bf16, (unsigned short)(p >> 16));
  }
  return a;
}
// B fragment 32x16 from transposed LDS tile [n][k]: col = lane&15,
// element e covers k = e + 16*(lane>=16) -> two contiguous 16B reads.
DEV v16bf load_b_frag_lds(const __bf16* bT, int col_local, int hi) {
  const __bf16* p = bT + col_local * LROW + hi * 16;
  v8bf lo = *(const v8bf*)(p);
  v8bf hi8 = *(const v8bf*)(p + 8);
  v16bf r;
#pragma unroll
  for (int e = 0; e < 8; ++e) { r[e] = lo[e]; r[8 + e] = hi8[e]; }
  return r;
}

DEV v8f wmma_bf16(v16bf a, v16bf b, v8f c) {
  return __builtin_amdgcn_wmma_f32_16x16x32_bf16(false, a, false, b, (short)0, c,
                                                 false, false);
}
DEV v8f v8f_zero() { v8f z = {0.f,0.f,0.f,0.f,0.f,0.f,0.f,0.f}; return z; }

// Stage a 32(K) x NC(N) f32 weight tile into a transposed, padded bf16 LDS
// tile [n][k]. Coalesced float4 reads, packed 2-k bf16 dword stores.
template <int NC>
DEV void stage_b_tile(const float* __restrict__ src, size_t ldn, int k0,
                      int colBase, int nValid, __bf16* dst, int tid) {
  constexpr int QU = NC / 4;
  for (int it = tid; it < 16 * QU; it += 256) {
    int kp = it / QU;
    int nq = (it - kp * QU) * 4;
    int gcol = colBase + nq;
    v4f a0 = {0.f,0.f,0.f,0.f}, a1 = {0.f,0.f,0.f,0.f};
    if (gcol < nValid) {
      const float* p = src + (size_t)(k0 + 2 * kp) * ldn + gcol;
      a0 = *(const v4f*)p;
      a1 = *(const v4f*)(p + ldn);
    }
#pragma unroll
    for (int i = 0; i < 4; ++i) {
      unsigned pk = ((unsigned)bfbits(a1[i]) << 16) | (unsigned)bfbits(a0[i]);
      *(unsigned*)((char*)dst + (nq + i) * (LROW * 2) + kp * 4) = pk;
    }
  }
}

// Wave 0 lanes async-copy the shared 16x32 bf16 A tile (two 16B chunks/lane).
DEV void stage_a_async(unsigned aOff, const __bf16* rowptr, int k0, int lane) {
  int c2 = lane & 1;
  unsigned lds = aOff + (unsigned)((lane >> 1) * (LROW * 2) + c2 * 32);
  async_copy_b128(lds, rowptr + k0 + c2 * 16);
  async_copy_b128(lds + 16, rowptr + k0 + c2 * 16 + 8);
}

// =====================================================================
// RMSNorm: f32 in -> bf16 out
// =====================================================================
__global__ void rmsnorm_kernel(const float* __restrict__ in,
                               const float* __restrict__ w,
                               __bf16* __restrict__ out, int C) {
  int row = blockIdx.x;
  const float* p = in + (size_t)row * C;
  __shared__ float red[256];
  float s = 0.f;
  for (int c = threadIdx.x; c < C; c += 256) { float v = p[c]; s += v * v; }
  red[threadIdx.x] = s;
  __syncthreads();
  for (int st = 128; st > 0; st >>= 1) {
    if (threadIdx.x < st) red[threadIdx.x] += red[threadIdx.x + st];
    __syncthreads();
  }
  float inv = rsqrtf(red[0] / (float)C + 1e-6f);
  for (int c = threadIdx.x; c < C; c += 256)
    out[(size_t)row * C + c] = f2bf(w[c] * (p[c] * inv));
}

// =====================================================================
// WMMA GEMM with LDS-staged tiles: C[M,N] = A_bf16[M,K] * Bw_f32[K,N] (+resid)
// block = 256 thr (8 waves); block tile 16(M) x 512(N); wave tile 16x64.
// A tile staged via gfx1250 async-to-LDS; B tile staged transposed in LDS.
// =====================================================================
template <bool RES, bool OUT_BF>
__global__ __launch_bounds__(256) void wmma_gemm_kernel(
    const __bf16* __restrict__ A, const float* __restrict__ Bw,
    const float* __restrict__ resid, float* __restrict__ Cf,
    __bf16* __restrict__ Cb, int M, int N, int K) {
  __shared__ __bf16 aT[16 * LROW];
  __shared__ __bf16 bT[512 * LROW];
  int tid = threadIdx.x;
  int lane = tid & 31, wave = tid >> 5;
  int hi = lane >> 4, ln = lane & 15;
  int row0 = blockIdx.y * 16;
  int col0 = blockIdx.x * 512;
  int wcol = wave * 64;
  unsigned aOff = (unsigned)(size_t)aT;
  const __bf16* aRow = A + (size_t)(row0 + (lane >> 1)) * K;  // wave-0 lanes

  v8f acc[4];
#pragma unroll
  for (int f = 0; f < 4; ++f) acc[f] = v8f_zero();

  for (int k0 = 0; k0 < K; k0 += 32) {
    if (wave == 0) stage_a_async(aOff, aRow, k0, lane);
    __builtin_prefetch(Bw + (size_t)(k0 + 32) * N + col0 + tid * 4, 0, 1);
    stage_b_tile<512>(Bw, (size_t)N, k0, col0, N, bT, tid);
    if (wave == 0) wait_asynccnt0();
    __syncthreads();
    v16bf a = load_a_frag_lds(aT, ln, hi);
#pragma unroll
    for (int f = 0; f < 4; ++f) {
      v16bf b = load_b_frag_lds(bT, wcol + f * 16 + ln, hi);
      acc[f] = wmma_bf16(a, b, acc[f]);
    }
    __syncthreads();
  }
#pragma unroll
  for (int f = 0; f < 4; ++f) {
    int col = col0 + wcol + f * 16 + ln;
#pragma unroll
    for (int j = 0; j < 8; ++j) {
      int r = row0 + j + hi * 8;
      float v = acc[f][j];
      if (RES) v += resid[(size_t)r * N + col];
      if (OUT_BF) Cb[(size_t)r * N + col] = f2bf(v);
      else        Cf[(size_t)r * N + col] = v;
    }
  }
}

// =====================================================================
// Per-head RMSNorm + RoPE for q and k, f32 in -> bf16 out
// =====================================================================
__global__ void qk_norm_rope_kernel(const float* __restrict__ qin,
                                    const float* __restrict__ kin,
                                    const float* __restrict__ qn_w,
                                    const float* __restrict__ kn_w,
                                    const float* __restrict__ cosb,
                                    const float* __restrict__ sinb,
                                    __bf16* __restrict__ qout,
                                    __bf16* __restrict__ kout) {
  int t = blockIdx.x, head = blockIdx.y, b = blockIdx.z;
  int d = threadIdx.x;
  bool isq = head < Hc;
  size_t tok = (size_t)b * Tc + t;
  const float* base;
  __bf16* obase;
  const float* nw;
  if (isq) {
    base = qin + (tok * Hc + head) * Dc;
    obase = qout + (tok * Hc + head) * Dc;
    nw = qn_w;
  } else {
    int kh = head - Hc;
    base = kin + (tok * KVc + kh) * Dc;
    obase = kout + (tok * KVc + kh) * Dc;
    nw = kn_w;
  }
  float x = base[d];
  __shared__ float red[128];
  red[d] = x * x;
  __syncthreads();
  for (int st = 64; st > 0; st >>= 1) {
    if (d < st) red[d] += red[d + st];
    __syncthreads();
  }
  float inv = rsqrtf(red[0] / (float)Dc + 1e-6f);
  float xn = nw[d] * (x * inv);
  __shared__ float xs[128];
  xs[d] = xn;
  __syncthreads();
  float rot = (d < Dc / 2) ? -xs[d + Dc / 2] : xs[d - Dc / 2];
  float c = cosb[tok * Dc + d];
  float s = sinb[tok * Dc + d];
  obase[d] = f2bf(xn * c + rot * s);
}

// =====================================================================
// Flash attention (causal, GQA). One wave per (b, h, 16-query tile).
// __launch_bounds__(32,1): full VGPR file for the single wave -> no spills.
// =====================================================================
__global__ __launch_bounds__(32, 1) void attn_kernel(
    const __bf16* __restrict__ q, const __bf16* __restrict__ k,
    const __bf16* __restrict__ v, __bf16* __restrict__ ao) {
  int qt = blockIdx.x, h = blockIdx.y, b = blockIdx.z;
  int lane = threadIdx.x;
  int hi = lane >> 4, ln = lane & 15;
  int kvh = h / (Hc / KVc);
  int qbase = qt * 16;
  __shared__ __bf16 pLDS[16 * 32];

  const __bf16* qptr = q + (((size_t)(b * Tc + qbase + ln) * Hc + h) * Dc);
  v16bf qa[4];
#pragma unroll
  for (int c = 0; c < 4; ++c) qa[c] = load_a_frag_g(qptr, c * 32, hi);

  float mrow[8], lrow[8];
  v8f oacc[8];
#pragma unroll
  for (int j = 0; j < 8; ++j) { mrow[j] = -3.0e38f; lrow[j] = 0.f; }
#pragma unroll
  for (int c = 0; c < 8; ++c) oacc[c] = v8f_zero();

  const float scale = 0.08838834764831845f;  // 1/sqrt(128)

  for (int kb = 0; kb < qbase + 16; kb += 32) {
    v8f s0 = v8f_zero(), s1 = v8f_zero();
#pragma unroll
    for (int c = 0; c < 4; ++c) {
      v16bf b0, b1;
      const __bf16* kp0 = k + (((size_t)(b * Tc + kb + ln) * KVc + kvh) * Dc + c * 32);
      const __bf16* kp1 = kp0 + (size_t)16 * KVc * Dc;
#pragma unroll
      for (int i = 0; i < 8; ++i) {
        int dl = 2 * i + hi * 16;
        unsigned p0 = *(const unsigned*)(kp0 + dl);
        unsigned p1 = *(const unsigned*)(kp1 + dl);
        b0[2 * i]     = __builtin_bit_cast(__bf16, (unsigned short)(p0 & 0xffffu));
        b0[2 * i + 1] = __builtin_bit_cast(__bf16, (unsigned short)(p0 >> 16));
        b1[2 * i]     = __builtin_bit_cast(__bf16, (unsigned short)(p1 & 0xffffu));
        b1[2 * i + 1] = __builtin_bit_cast(__bf16, (unsigned short)(p1 >> 16));
      }
      s0 = wmma_bf16(qa[c], b0, s0);
      s1 = wmma_bf16(qa[c], b1, s1);
    }
#pragma unroll
    for (int j = 0; j < 8; ++j) {
      int qi = qbase + j + hi * 8;
      float v0 = s0[j] * scale;
      float v1 = s1[j] * scale;
      if (kb + ln > qi)      v0 = -1.0e30f;
      if (kb + 16 + ln > qi) v1 = -1.0e30f;
      float tmax = fmaxf(v0, v1);
#pragma unroll
      for (int m = 1; m < 16; m <<= 1) tmax = fmaxf(tmax, __shfl_xor(tmax, m, 32));
      float mnew = fmaxf(mrow[j], tmax);
      float sc = __expf(mrow[j] - mnew);
      float p0 = __expf(v0 - mnew);
      float p1 = __expf(v1 - mnew);
      float rs = p0 + p1;
#pragma unroll
      for (int m = 1; m < 16; m <<= 1) rs += __shfl_xor(rs, m, 32);
      lrow[j] = lrow[j] * sc + rs;
      mrow[j] = mnew;
#pragma unroll
      for (int c = 0; c < 8; ++c) oacc[c][j] *= sc;
      int r = j + hi * 8;
      pLDS[r * 32 + ln]      = f2bf(p0);
      pLDS[r * 32 + 16 + ln] = f2bf(p1);
    }
    __syncthreads();
    v16bf pa;
#pragma unroll
    for (int i = 0; i < 8; ++i) {
      int kbp = 2 * i + hi * 16;
      unsigned pp = *(const unsigned*)(&pLDS[ln * 32 + kbp]);
      pa[2 * i]     = __builtin_bit_cast(__bf16, (unsigned short)(pp & 0xffffu));
      pa[2 * i + 1] = __builtin_bit_cast(__bf16, (unsigned short)(pp >> 16));
    }
#pragma unroll
    for (int c = 0; c < 8; ++c) {
      v16bf bv;
#pragma unroll
      for (int e = 0; e < 16; ++e) {
        int kk = e + hi * 16;
        bv[e] = v[((size_t)(b * Tc + kb + kk) * KVc + kvh) * Dc + c * 16 + ln];
      }
      oacc[c] = wmma_bf16(pa, bv, oacc[c]);
    }
    __syncthreads();
  }
#pragma unroll
  for (int c = 0; c < 8; ++c)
#pragma unroll
    for (int j = 0; j < 8; ++j) {
      int r = qbase + j + hi * 8;
      float o = oacc[c][j] / lrow[j];
      ao[((size_t)(b * Tc + r) * Hc + h) * Dc + c * 16 + ln] = f2bf(o);
    }
}

// =====================================================================
// Router: softmax -> top-2 -> normalized weights, scatter to expert lists
// =====================================================================
__global__ void router_kernel(const __bf16* __restrict__ h2,
                              const float* __restrict__ gate_w,
                              int* __restrict__ count,
                              int* __restrict__ elist,
                              float* __restrict__ wlist) {
  int t = blockIdx.x;
  int tid = threadIdx.x;  // 128
  float acc[Ec];
#pragma unroll
  for (int e = 0; e < Ec; ++e) acc[e] = 0.f;
  for (int c = tid; c < Cc; c += 128) {
    float hv = bf2f(h2[(size_t)t * Cc + c]);
#pragma unroll
    for (int e = 0; e < Ec; ++e) acc[e] += hv * gate_w[(size_t)c * Ec + e];
  }
  __shared__ float red[Ec * 128];
#pragma unroll
  for (int e = 0; e < Ec; ++e) red[e * 128 + tid] = acc[e];
  __syncthreads();
  for (int st = 64; st > 0; st >>= 1) {
    if (tid < st)
#pragma unroll
      for (int e = 0; e < Ec; ++e) red[e * 128 + tid] += red[e * 128 + tid + st];
    __syncthreads();
  }
  if (tid == 0) {
    float lg[Ec], sm[Ec];
    float mx = -3.0e38f;
#pragma unroll
    for (int e = 0; e < Ec; ++e) { lg[e] = red[e * 128]; mx = fmaxf(mx, lg[e]); }
    float ssum = 0.f;
#pragma unroll
    for (int e = 0; e < Ec; ++e) { sm[e] = __expf(lg[e] - mx); ssum += sm[e]; }
#pragma unroll
    for (int e = 0; e < Ec; ++e) sm[e] /= ssum;
    int i1 = 0;
#pragma unroll
    for (int e = 1; e < Ec; ++e) if (sm[e] > sm[i1]) i1 = e;
    int i2 = (i1 == 0) ? 1 : 0;
#pragma unroll
    for (int e = 0; e < Ec; ++e) if (e != i1 && sm[e] > sm[i2]) i2 = e;
    float w1 = sm[i1], w2 = sm[i2];
    float s = w1 + w2 + 1e-9f;
    w1 /= s; w2 /= s;
    int p1 = atomicAdd(&count[i1], 1);
    elist[i1 * NTOK + p1] = t; wlist[i1 * NTOK + p1] = w1;
    int p2 = atomicAdd(&count[i2], 1);
    elist[i2 * NTOK + p2] = t; wlist[i2 * NTOK + p2] = w2;
  }
}

// =====================================================================
// MoE gate_up: act = silu(g)*u bf16, routed tokens only (gathered rows).
// grid = (ceil(F/512), NTOK/16, E), block = 256
// =====================================================================
__global__ __launch_bounds__(256) void moe_gate_up_kernel(
    const __bf16* __restrict__ h2, const float* __restrict__ gup,
    const int* __restrict__ count, const int* __restrict__ elist,
    __bf16* __restrict__ act) {
  __shared__ __bf16 aT[16 * LROW];
  __shared__ __bf16 bTg[512 * LROW];
  __shared__ __bf16 bTu[512 * LROW];
  int e = blockIdx.z;
  int cnt = count[e];
  int tile = blockIdx.y;
  if (tile * 16 >= cnt) return;
  int tid = threadIdx.x;
  int lane = tid & 31, wave = tid >> 5;
  int hi = lane >> 4, ln = lane & 15;
  int col0 = blockIdx.x * 512;
  int wcol = wave * 64;
  bool wvalid = (col0 + wcol) < Fc;
  unsigned aOff = (unsigned)(size_t)aT;
  int slotA = tile * 16 + (lane >> 1);
  int tokA = (slotA < cnt) ? elist[e * NTOK + slotA] : 0;
  const __bf16* aRow = h2 + (size_t)tokA * Cc;
  const float* bbase = gup + (size_t)e * Cc * (2 * Fc);

  v8f accG[4], accU[4];
#pragma unroll
  for (int f = 0; f < 4; ++f) { accG[f] = v8f_zero(); accU[f] = v8f_zero(); }

  for (int k0 = 0; k0 < Cc; k0 += 32) {
    if (wave == 0) stage_a_async(aOff, aRow, k0, lane);
    stage_b_tile<512>(bbase, (size_t)(2 * Fc), k0, col0, Fc, bTg, tid);
    stage_b_tile<512>(bbase + Fc, (size_t)(2 * Fc), k0, col0, Fc, bTu, tid);
    if (wave == 0) wait_asynccnt0();
    __syncthreads();
    if (wvalid) {
      v16bf a = load_a_frag_lds(aT, ln, hi);
#pragma unroll
      for (int f = 0; f < 4; ++f) {
        v16bf bg = load_b_frag_lds(bTg, wcol + f * 16 + ln, hi);
        v16bf bu = load_b_frag_lds(bTu, wcol + f * 16 + ln, hi);
        accG[f] = wmma_bf16(a, bg, accG[f]);
        accU[f] = wmma_bf16(a, bu, accU[f]);
      }
    }
    __syncthreads();
  }
  if (!wvalid) return;
#pragma unroll
  for (int f = 0; f < 4; ++f) {
    int col = col0 + wcol + f * 16 + ln;
    if (col >= Fc) continue;
#pragma unroll
    for (int j = 0; j < 8; ++j) {
      int rslot = tile * 16 + j + hi * 8;
      if (rslot < cnt) {
        float g = accG[f][j], u = accU[f][j];
        float a = (g / (1.f + __expf(-g))) * u;  // silu(g)*u
        act[((size_t)e * NTOK + rslot) * Fc + col] = f2bf(a);
      }
    }
  }
}

// =====================================================================
// MoE down: out[token] += w * (act @ down_w[e]) via f32 global atomics.
// grid = (C/512, NTOK/16, E), block = 256
// =====================================================================
__global__ __launch_bounds__(256) void moe_down_kernel(
    const __bf16* __restrict__ act, const float* __restrict__ dw,
    const int* __restrict__ count, const int* __restrict__ elist,
    const float* __restrict__ wlist, float* __restrict__ out) {
  __shared__ __bf16 aT[16 * LROW];
  __shared__ __bf16 bT[512 * LROW];
  int e = blockIdx.z;
  int cnt = count[e];
  int tile = blockIdx.y;
  if (tile * 16 >= cnt) return;
  int tid = threadIdx.x;
  int lane = tid & 31, wave = tid >> 5;
  int hi = lane >> 4, ln = lane & 15;
  int col0 = blockIdx.x * 512;
  int wcol = wave * 64;
  unsigned aOff = (unsigned)(size_t)aT;
  int slotA = tile * 16 + (lane >> 1);
  int clampA = (slotA < cnt) ? slotA : (cnt - 1);
  const __bf16* aRow = act + ((size_t)e * NTOK + clampA) * Fc;
  const float* bbase = dw + (size_t)e * Fc * Cc;

  v8f acc[4];
#pragma unroll
  for (int f = 0; f < 4; ++f) acc[f] = v8f_zero();

  for (int k0 = 0; k0 < Fc; k0 += 32) {
    if (wave == 0) stage_a_async(aOff, aRow, k0, lane);
    stage_b_tile<512>(bbase, (size_t)Cc, k0, col0, Cc, bT, tid);
    if (wave == 0) wait_asynccnt0();
    __syncthreads();
    v16bf a = load_a_frag_lds(aT, ln, hi);
#pragma unroll
    for (int f = 0; f < 4; ++f) {
      v16bf b = load_b_frag_lds(bT, wcol + f * 16 + ln, hi);
      acc[f] = wmma_bf16(a, b, acc[f]);
    }
    __syncthreads();
  }
#pragma unroll
  for (int f = 0; f < 4; ++f) {
    int col = col0 + wcol + f * 16 + ln;
#pragma unroll
    for (int j = 0; j < 8; ++j) {
      int rslot = tile * 16 + j + hi * 8;
      if (rslot < cnt) {
        int tok = elist[e * NTOK + rslot];
        float w = wlist[e * NTOK + rslot];
        atomicAdd(&out[(size_t)tok * Cc + col], w * acc[f][j]);
      }
    }
  }
}

// =====================================================================
// Launcher
// =====================================================================
extern "C" void kernel_launch(void* const* d_in, const int* in_sizes, int n_in,
                              void* d_out, int out_size, void* d_ws, size_t ws_size,
                              hipStream_t stream) {
  const float* x      = (const float*)d_in[0];
  const float* cosb   = (const float*)d_in[1];
  const float* sinb   = (const float*)d_in[2];
  const float* ln1_w  = (const float*)d_in[3];
  const float* q_w    = (const float*)d_in[4];
  const float* k_w    = (const float*)d_in[5];
  const float* v_w    = (const float*)d_in[6];
  const float* o_w    = (const float*)d_in[7];
  const float* qn_w   = (const float*)d_in[8];
  const float* kn_w   = (const float*)d_in[9];
  const float* ln2_w  = (const float*)d_in[10];
  const float* gate_w = (const float*)d_in[11];
  const float* gup_w  = (const float*)d_in[12];
  const float* down_w = (const float*)d_in[13];
  float* out = (float*)d_out;

  char* ws = (char*)d_ws;
  size_t off = 0;
  auto alloc = [&](size_t bytes) -> char* {
    char* p = ws + off;
    off = (off + bytes + 255) & ~(size_t)255;
    return p;
  };
  __bf16* h_bf = (__bf16*)alloc((size_t)NTOK * Cc * 2);
  float*  qbuf = (float*)alloc((size_t)NTOK * Hc * Dc * 4);
  float*  kbuf = (float*)alloc((size_t)NTOK * KVc * Dc * 4);
  __bf16* vr   = (__bf16*)alloc((size_t)NTOK * KVc * Dc * 2);
  __bf16* qr   = (__bf16*)alloc((size_t)NTOK * Hc * Dc * 2);
  __bf16* kr   = (__bf16*)alloc((size_t)NTOK * KVc * Dc * 2);
  __bf16* ao   = (__bf16*)alloc((size_t)NTOK * Hc * Dc * 2);
  __bf16* h2   = (__bf16*)alloc((size_t)NTOK * Cc * 2);
  int*    count = (int*)alloc(Ec * 4);
  int*    elist = (int*)alloc((size_t)Ec * NTOK * 4);
  float*  wlist = (float*)alloc((size_t)Ec * NTOK * 4);
  __bf16* act   = (__bf16*)alloc((size_t)Ec * NTOK * Fc * 2);

  rmsnorm_kernel<<<NTOK, 256, 0, stream>>>(x, ln1_w, h_bf, Cc);

  wmma_gemm_kernel<false, false><<<dim3(8, NTOK / 16), 256, 0, stream>>>(
      h_bf, q_w, nullptr, qbuf, nullptr, NTOK, Hc * Dc, Cc);
  wmma_gemm_kernel<false, false><<<dim3(2, NTOK / 16), 256, 0, stream>>>(
      h_bf, k_w, nullptr, kbuf, nullptr, NTOK, KVc * Dc, Cc);
  wmma_gemm_kernel<false, true><<<dim3(2, NTOK / 16), 256, 0, stream>>>(
      h_bf, v_w, nullptr, nullptr, vr, NTOK, KVc * Dc, Cc);

  qk_norm_rope_kernel<<<dim3(Tc, Hc + KVc, Bc), 128, 0, stream>>>(
      qbuf, kbuf, qn_w, kn_w, cosb, sinb, qr, kr);

  attn_kernel<<<dim3(Tc / 16, Hc, Bc), 32, 0, stream>>>(qr, kr, vr, ao);

  wmma_gemm_kernel<true, false><<<dim3(4, NTOK / 16), 256, 0, stream>>>(
      ao, o_w, x, out, nullptr, NTOK, Cc, Hc * Dc);

  rmsnorm_kernel<<<NTOK, 256, 0, stream>>>(out, ln2_w, h2, Cc);

  hipMemsetAsync(count, 0, Ec * 4, stream);
  router_kernel<<<NTOK, 128, 0, stream>>>(h2, gate_w, count, elist, wlist);

  moe_gate_up_kernel<<<dim3(2, NTOK / 16, Ec), 256, 0, stream>>>(
      h2, gup_w, count, elist, act);
  moe_down_kernel<<<dim3(4, NTOK / 16, Ec), 256, 0, stream>>>(
      act, down_w, count, elist, wlist, out);
}
